// BiDirectionalMambaBlock_70480413327619
// MI455X (gfx1250) — compile-verified, matrix operand order checked by reading, not run on hardware
//
#include <hip/hip_runtime.h>
#include <hip/hip_bf16.h>

// ---------------------------------------------------------------- constants
#define D_MODEL  1024
#define D_INNER  2048
#define D_CONV   4
#define D_STATE  16
#define DT_RANK  64
#define BATCH    2
#define SEQ      1024
#define MROWS    (BATCH * SEQ)           // 2048
#define NX       (DT_RANK + 2 * D_STATE) // 96

typedef __attribute__((ext_vector_type(16))) __bf16 bf16x16;
typedef __attribute__((ext_vector_type(8)))  __bf16 bf16x8;
typedef __attribute__((ext_vector_type(8)))  float  f32x8;
typedef __attribute__((ext_vector_type(4)))  int    v4i;

// ---------------------------------------------------------------- gfx1250 async global->LDS helpers
__device__ __forceinline__ void async_copy_16B(const __bf16* g, __bf16* l) {
#if __has_builtin(__builtin_amdgcn_global_load_async_to_lds_b128)
    __builtin_amdgcn_global_load_async_to_lds_b128((v4i*)g, (v4i*)l, 0, 0);
#else
    *(bf16x8*)l = *(const bf16x8*)g;   // synchronous fallback
#endif
}

__device__ __forceinline__ void async_wait0() {
#if __has_builtin(__builtin_amdgcn_s_wait_asynccnt)
    __builtin_amdgcn_s_wait_asynccnt(0);
#else
    asm volatile("s_wait_asynccnt 0x0" ::: "memory");
#endif
}

// ---------------------------------------------------------------- cast f32 -> bf16 (strided source)
__global__ void cast_f32_bf16(const float* __restrict__ src, int srcStride,
                              __bf16* __restrict__ dst, int rows, int cols) {
    int idx = blockIdx.x * blockDim.x + threadIdx.x;
    if (idx >= rows * cols) return;
    int r = idx / cols;
    int c = idx - r * cols;
    dst[idx] = (__bf16)src[(size_t)r * srcStride + c];
}

// ---------------------------------------------------------------- LDS-tiled, double-buffered WMMA GEMM
// C[M,N] = A[M,K] * B[N,K]^T.  Block = 256 threads (8 waves), tile 128(M) x 128(N), BK=64.
// Each wave computes 32(M) x 64(N): 2 M sub-tiles x 4 N sub-tiles = 8 accumulators.
// Global->LDS via async copies (ASYNCcnt), overlapped with WMMA on the other buffer.
#define BM 128
#define BN 128
#define BK 64

__global__ __launch_bounds__(256)
void gemm_bf16_wmma_lds(const __bf16* __restrict__ A, int lda,
                        const __bf16* __restrict__ B, int ldb,
                        float* __restrict__ C, int ldc, int K) {
    __shared__ __align__(16) __bf16 sA[2][BM * BK];   // 32 KB
    __shared__ __align__(16) __bf16 sB[2][BN * BK];   // 32 KB

    const int tid   = threadIdx.x;
    const int lane  = tid & 31;
    const int wid   = tid >> 5;          // 0..7
    const int waveM = wid & 3;           // 4 M sub-groups of 32
    const int waveN = wid >> 2;          // 2 N sub-groups of 64
    const int r     = lane & 15;
    const int kOff  = (lane >> 4) * 8;   // ISA 16-bit A/B lane layout
    const int mBase = blockIdx.y * BM;
    const int nBase = blockIdx.x * BN;

    f32x8 acc[2][4] = {};

    // (128+128) rows * 64 cols * 2B / 16B = 2048 chunks; 8 per thread
    auto stage = [&](int buf, int kStart) {
#pragma unroll
        for (int i = 0; i < 8; ++i) {
            int c = tid + i * 256;
            if (c < 1024) {
                int row = c >> 3, col = (c & 7) * 8;
                async_copy_16B(A + (size_t)(mBase + row) * lda + kStart + col,
                               &sA[buf][row * BK + col]);
            } else {
                int cb  = c - 1024;
                int row = cb >> 3, col = (cb & 7) * 8;
                async_copy_16B(B + (size_t)(nBase + row) * ldb + kStart + col,
                               &sB[buf][row * BK + col]);
            }
        }
    };

    stage(0, 0);
    const int nk = K / BK;
    for (int kb = 0; kb < nk; ++kb) {
        const int cur = kb & 1;
        async_wait0();                    // my async writes to LDS have landed
        __syncthreads();                  // everyone's landed; prev compute done
        if (kb + 1 < nk) stage(cur ^ 1, (kb + 1) * BK);   // overlap with WMMA below
#pragma unroll
        for (int ks = 0; ks < BK; ks += 32) {
            // load ALL fragments of this k-step first, then 8 WMMAs back-to-back
            union { bf16x16 v; bf16x8 h[2]; } af[2], bfm[4];
#pragma unroll
            for (int m = 0; m < 2; ++m) {
                const __bf16* ap =
                    &sA[cur][(waveM * 32 + m * 16 + r) * BK + ks + kOff];
                af[m].h[0] = *(const bf16x8*)ap;
                af[m].h[1] = *(const bf16x8*)(ap + 16);
            }
#pragma unroll
            for (int j = 0; j < 4; ++j) {
                const __bf16* bp =
                    &sB[cur][(waveN * 64 + j * 16 + r) * BK + ks + kOff];
                bfm[j].h[0] = *(const bf16x8*)bp;
                bfm[j].h[1] = *(const bf16x8*)(bp + 16);
            }
#pragma unroll
            for (int j = 0; j < 4; ++j)
#pragma unroll
                for (int m = 0; m < 2; ++m)
                    acc[m][j] = __builtin_amdgcn_wmma_f32_16x16x32_bf16(
                        false, af[m].v, false, bfm[j].v, (short)0, acc[m][j],
                        false, false);
        }
    }

    // C/D layout: VGPR i -> row +i (+8 for lanes 16-31), col = lane&15
    const int rowHi = (lane >> 4) * 8;
#pragma unroll
    for (int m = 0; m < 2; ++m) {
        const int mRow = mBase + waveM * 32 + m * 16 + rowHi;
#pragma unroll
        for (int j = 0; j < 4; ++j) {
            const int n = nBase + waveN * 64 + j * 16 + r;
#pragma unroll
            for (int i = 0; i < 8; ++i)
                C[(size_t)(mRow + i) * ldc + n] = acc[m][j][i];
        }
    }
}

// ---------------------------------------------------------------- direct WMMA GEMM (skinny N=96 path)
template <int NT>
__global__ __launch_bounds__(32)
void gemm_bf16_wmma(const __bf16* __restrict__ A, int lda,
                    const __bf16* __restrict__ B, int ldb,
                    float* __restrict__ C, int ldc, int K) {
    const int lane  = threadIdx.x;
    const int r     = lane & 15;
    const int kOff  = (lane >> 4) * 8;
    const int mBase = blockIdx.y * 16;
    const int nBase = blockIdx.x * (16 * NT);

    const __bf16* aRow = A + (size_t)(mBase + r) * lda + kOff;

    f32x8 acc[NT] = {};
    for (int k = 0; k < K; k += 32) {
        union { bf16x16 v; bf16x8 h[2]; } af;
        af.h[0] = *(const bf16x8*)(aRow + k);
        af.h[1] = *(const bf16x8*)(aRow + k + 16);
        __builtin_prefetch(aRow + k + 32, 0, 1);
#pragma unroll
        for (int j = 0; j < NT; ++j) {
            const __bf16* bRow = B + (size_t)(nBase + j * 16 + r) * ldb + kOff;
            union { bf16x16 v; bf16x8 h[2]; } bfm;
            bfm.h[0] = *(const bf16x8*)(bRow + k);
            bfm.h[1] = *(const bf16x8*)(bRow + k + 16);
            acc[j] = __builtin_amdgcn_wmma_f32_16x16x32_bf16(
                false, af.v, false, bfm.v, (short)0, acc[j], false, false);
        }
    }
    const int rowHi = (lane >> 4) * 8;
#pragma unroll
    for (int j = 0; j < NT; ++j) {
        const int n = nBase + j * 16 + r;
#pragma unroll
        for (int i = 0; i < 8; ++i)
            C[(size_t)(mBase + rowHi + i) * ldc + n] = acc[j][i];
    }
}

// ---------------------------------------------------------------- depthwise causal conv + SiLU
__global__ void conv_silu(const float* __restrict__ xr,
                          const float* __restrict__ wconv,
                          const float* __restrict__ bconv,
                          float* __restrict__ u, __bf16* __restrict__ u_bf) {
    int idx = blockIdx.x * blockDim.x + threadIdx.x;
    if (idx >= MROWS * D_INNER) return;
    int row = idx >> 11;
    int d   = idx & (D_INNER - 1);
    int l   = row & (SEQ - 1);
    float acc = bconv[d];
#pragma unroll
    for (int j = 0; j < D_CONV; ++j) {
        int ls = l + j - (D_CONV - 1);
        if (ls >= 0)
            acc += wconv[d * D_CONV + j] *
                   xr[(size_t)(row + j - (D_CONV - 1)) * (2 * D_INNER) + d];
    }
    float s = acc / (1.0f + __expf(-acc));
    u[idx]    = s;
    u_bf[idx] = (__bf16)s;
}

// ---------------------------------------------------------------- softplus(dp + bdt) in-place
__global__ void softplus_bias(float* __restrict__ dp, const float* __restrict__ bdt) {
    int idx = blockIdx.x * blockDim.x + threadIdx.x;
    if (idx >= MROWS * D_INNER) return;
    int d = idx & (D_INNER - 1);
    float x = dp[idx] + bdt[d];
    dp[idx] = (x > 20.0f) ? x : log1pf(__expf(x));
}

// ---------------------------------------------------------------- selective scan (one direction)
__global__ __launch_bounds__(256)
void s6_scan(const float* __restrict__ u, const float* __restrict__ delta,
             const float* __restrict__ dbc, const float* __restrict__ A_log,
             const float* __restrict__ Dp, float* __restrict__ y, int reverse) {
    __shared__ float sB[D_STATE];
    __shared__ float sC[D_STATE];
    const int b = blockIdx.y;
    const int d = blockIdx.x * blockDim.x + threadIdx.x;

    float A[D_STATE], h[D_STATE];
#pragma unroll
    for (int n = 0; n < D_STATE; ++n) {
        A[n] = -__expf(A_log[d * D_STATE + n]);
        h[n] = 0.0f;
    }
    const float Dv = Dp[d];

    for (int t = 0; t < SEQ; ++t) {
        const int l   = reverse ? (SEQ - 1 - t) : t;
        const int row = b * SEQ + l;
        if (threadIdx.x < 2 * D_STATE) {
            float v = dbc[(size_t)row * NX + DT_RANK + threadIdx.x];
            if (threadIdx.x < D_STATE) sB[threadIdx.x] = v;
            else                       sC[threadIdx.x - D_STATE] = v;
        }
        __syncthreads();
        const float del = delta[(size_t)row * D_INNER + d];
        const float uv  = u[(size_t)row * D_INNER + d];
        const float du  = del * uv;
        float acc = 0.0f;
#pragma unroll
        for (int n = 0; n < D_STATE; ++n) {
            float dA = __expf(del * A[n]);
            h[n] = dA * h[n] + du * sB[n];
            acc += h[n] * sC[n];
        }
        y[(size_t)row * D_INNER + d] = acc + uv * Dv;
        __syncthreads();
    }
}

// ---------------------------------------------------------------- combine + gate, emit bf16
__global__ void combine_gate(const float* __restrict__ yf, const float* __restrict__ yb,
                             const float* __restrict__ xr, __bf16* __restrict__ ybf) {
    int idx = blockIdx.x * blockDim.x + threadIdx.x;
    if (idx >= MROWS * D_INNER) return;
    int row = idx >> 11;
    int d   = idx & (D_INNER - 1);
    float res  = xr[(size_t)row * (2 * D_INNER) + D_INNER + d];
    float sres = res / (1.0f + __expf(-res));
    float v = (yf[idx] + yb[idx]) * 0.5f * sres;
    ybf[idx] = (__bf16)v;
}

// ---------------------------------------------------------------- launcher
extern "C" void kernel_launch(void* const* d_in, const int* in_sizes, int n_in,
                              void* d_out, int out_size, void* d_ws, size_t ws_size,
                              hipStream_t stream) {
    (void)in_sizes; (void)n_in; (void)out_size; (void)ws_size;
    const float* x      = (const float*)d_in[0];
    const float* W_in   = (const float*)d_in[1];
    const float* W_conv = (const float*)d_in[2];
    const float* b_conv = (const float*)d_in[3];
    const float* W_out  = (const float*)d_in[4];
    const float* fA_log = (const float*)d_in[5];
    const float* fD     = (const float*)d_in[6];
    const float* fWx    = (const float*)d_in[7];
    const float* fWdt   = (const float*)d_in[8];
    const float* fbdt   = (const float*)d_in[9];
    const float* bA_log = (const float*)d_in[10];
    const float* bD     = (const float*)d_in[11];
    const float* bWx    = (const float*)d_in[12];
    const float* bWdt   = (const float*)d_in[13];
    const float* bbdt   = (const float*)d_in[14];
    float* out = (float*)d_out;

    char* ws = (char*)d_ws;
    size_t off = 0;
    auto alloc = [&](size_t bytes) -> void* {
        off = (off + 255) & ~(size_t)255;
        void* p = ws + off;
        off += bytes;
        return p;
    };

    float*  xr      = (float*) alloc((size_t)MROWS * 2 * D_INNER * 4);
    __bf16* x_bf    = (__bf16*)alloc((size_t)MROWS * D_MODEL * 2);
    __bf16* Win_bf  = (__bf16*)alloc((size_t)2 * D_INNER * D_MODEL * 2);
    float*  u       = (float*) alloc((size_t)MROWS * D_INNER * 4);
    __bf16* u_bf    = (__bf16*)alloc((size_t)MROWS * D_INNER * 2);
    __bf16* fWx_bf  = (__bf16*)alloc((size_t)NX * D_INNER * 2);
    __bf16* bWx_bf  = (__bf16*)alloc((size_t)NX * D_INNER * 2);
    __bf16* fWdt_bf = (__bf16*)alloc((size_t)D_INNER * DT_RANK * 2);
    __bf16* bWdt_bf = (__bf16*)alloc((size_t)D_INNER * DT_RANK * 2);
    __bf16* Wout_bf = (__bf16*)alloc((size_t)D_MODEL * D_INNER * 2);
    float*  dbc_f   = (float*) alloc((size_t)MROWS * NX * 4);
    float*  dbc_b   = (float*) alloc((size_t)MROWS * NX * 4);
    __bf16* dt_bf   = (__bf16*)alloc((size_t)MROWS * DT_RANK * 2);
    float*  delta_f = (float*) alloc((size_t)MROWS * D_INNER * 4);
    float*  delta_b = (float*) alloc((size_t)MROWS * D_INNER * 4);
    float*  y_f     = (float*) alloc((size_t)MROWS * D_INNER * 4);
    float*  y_b     = (float*) alloc((size_t)MROWS * D_INNER * 4);
    __bf16* y_bf    = (__bf16*)alloc((size_t)MROWS * D_INNER * 2);

    auto cast = [&](const float* src, int stride, __bf16* dst, int rows, int cols) {
        int total = rows * cols;
        cast_f32_bf16<<<(total + 255) / 256, 256, 0, stream>>>(src, stride, dst, rows, cols);
    };

    // ---- bf16 weight/activation prep
    cast(x,     D_MODEL, x_bf,    MROWS,       D_MODEL);
    cast(W_in,  D_MODEL, Win_bf,  2 * D_INNER, D_MODEL);
    cast(fWx,   D_INNER, fWx_bf,  NX,          D_INNER);
    cast(bWx,   D_INNER, bWx_bf,  NX,          D_INNER);
    cast(fWdt,  DT_RANK, fWdt_bf, D_INNER,     DT_RANK);
    cast(bWdt,  DT_RANK, bWdt_bf, D_INNER,     DT_RANK);
    cast(W_out, D_INNER, Wout_bf, D_MODEL,     D_INNER);

    const int ew_total  = MROWS * D_INNER;
    const int ew_blocks = (ew_total + 255) / 256;

    // ---- in-projection: xr = x @ W_in^T   (2048 x 4096 x 1024) -- LDS-tiled WMMA
    gemm_bf16_wmma_lds<<<dim3((2 * D_INNER) / BN, MROWS / BM), 256, 0, stream>>>(
        x_bf, D_MODEL, Win_bf, D_MODEL, xr, 2 * D_INNER, D_MODEL);

    // ---- depthwise causal conv + SiLU
    conv_silu<<<ew_blocks, 256, 0, stream>>>(xr, W_conv, b_conv, u, u_bf);

    // ---- forward direction projections
    gemm_bf16_wmma<6><<<dim3(NX / 96, MROWS / 16), 32, 0, stream>>>(
        u_bf, D_INNER, fWx_bf, D_INNER, dbc_f, NX, D_INNER);
    cast(dbc_f, NX, dt_bf, MROWS, DT_RANK);
    gemm_bf16_wmma_lds<<<dim3(D_INNER / BN, MROWS / BM), 256, 0, stream>>>(
        dt_bf, DT_RANK, fWdt_bf, DT_RANK, delta_f, D_INNER, DT_RANK);
    softplus_bias<<<ew_blocks, 256, 0, stream>>>(delta_f, fbdt);

    // ---- backward direction projections
    gemm_bf16_wmma<6><<<dim3(NX / 96, MROWS / 16), 32, 0, stream>>>(
        u_bf, D_INNER, bWx_bf, D_INNER, dbc_b, NX, D_INNER);
    cast(dbc_b, NX, dt_bf, MROWS, DT_RANK);
    gemm_bf16_wmma_lds<<<dim3(D_INNER / BN, MROWS / BM), 256, 0, stream>>>(
        dt_bf, DT_RANK, bWdt_bf, DT_RANK, delta_b, D_INNER, DT_RANK);
    softplus_bias<<<ew_blocks, 256, 0, stream>>>(delta_b, bbdt);

    // ---- selective scans: fwd then bwd
    s6_scan<<<dim3(D_INNER / 256, BATCH), 256, 0, stream>>>(
        u, delta_f, dbc_f, fA_log, fD, y_f, 0);
    s6_scan<<<dim3(D_INNER / 256, BATCH), 256, 0, stream>>>(
        u, delta_b, dbc_b, bA_log, bD, y_b, 1);

    // ---- combine + SiLU gate -> bf16
    combine_gate<<<ew_blocks, 256, 0, stream>>>(y_f, y_b, xr, y_bf);

    // ---- out-projection: out = y @ W_out^T   (2048 x 1024 x 2048) -- LDS-tiled WMMA
    gemm_bf16_wmma_lds<<<dim3(D_MODEL / BN, MROWS / BM), 256, 0, stream>>>(
        y_bf, D_INNER, Wout_bf, D_INNER, out, D_MODEL, D_INNER);
}
